// Unet_59966333386782
// MI455X (gfx1250) — matrix operand order repaired and checked by reading, not compiled, not run on hardware
//
#include <hip/hip_runtime.h>
#include <hip/hip_bf16.h>

typedef __attribute__((ext_vector_type(16))) __bf16 v16bf;
typedef __attribute__((ext_vector_type(8)))  __bf16 v8bf;
typedef __attribute__((ext_vector_type(8)))  float  v8f;

// Problem constants (from reference)
constexpr int BB = 16, C = 128, H = 56, W = 56;
constexpr int QK = 128, HEADS = 4, NWIN = 7, TOPK = 4;
constexpr int HD = 32;               // HD_Q == HD_V == 32
constexpr int WH = 8, P2 = 49, W2 = 64;
constexpr int TOK = BB * H * W;      // 50176
constexpr int NQKV = 2 * QK + C;     // 384
constexpr float SCALE = 0.08838834764831845f;  // 128^-0.5

// ---------------- WMMA fragment helpers (wave32, 16x16x32 bf16) ----------------
__device__ __forceinline__ v16bf cat8(v8bf lo, v8bf hi) {
    return __builtin_shufflevector(lo, hi, 0, 1, 2, 3, 4, 5, 6, 7, 8, 9, 10, 11, 12, 13, 14, 15);
}
// A (16x32 row-major, lda elems): lane<16 -> row=lane, K = kb..kb+7 then kb+16..kb+23 (kb=0)
//                                 lane>=16 -> row=lane-16, kb=8
__device__ __forceinline__ v16bf frag_a_rm(const __bf16* A, int lda, int row0, int k0) {
    int lane = threadIdx.x & 31;
    const __bf16* p = A + (size_t)(row0 + (lane & 15)) * lda + k0 + ((lane >> 4) << 3);
    v8bf lo = *(const v8bf*)p;
    v8bf hi = *(const v8bf*)(p + 16);
    return cat8(lo, hi);
}
// B given TRANSPOSED storage Bt[N][K] (ldk elems): B[k][n] = Bt[n*ldk + k]
// lane<16 -> col=lane, K = k0..k0+7 then k0+16..k0+23 ; lane>=16 -> col=lane-16, K = k0+8..k0+15 then k0+24..k0+31
__device__ __forceinline__ v16bf frag_b_t(const __bf16* Bt, int ldk, int k0, int col0) {
    int lane = threadIdx.x & 31;
    const __bf16* p = Bt + (size_t)(col0 + (lane & 15)) * ldk + k0 + ((lane >> 4) << 3);
    v8bf lo = *(const v8bf*)p;
    v8bf hi = *(const v8bf*)(p + 16);
    return cat8(lo, hi);
}

// ---------------- weight convert + transpose: in[K,N] fp32 -> out[N,K] bf16 ----------------
__global__ void f2bf_t_kernel(const float* __restrict__ in, __bf16* __restrict__ out, int K, int N) {
    int i = blockIdx.x * blockDim.x + threadIdx.x;
    if (i >= K * N) return;
    int k = i / N, n = i - k * N;
    out[(size_t)n * K + k] = (__bf16)in[i];
}

// x NCHW -> xh NHWC with 3x3 depthwise conv residual: xh = x + dwconv3(x) + pos_b
__global__ void posconv_kernel(const float* __restrict__ x, const float* __restrict__ wt,
                               const float* __restrict__ bias, float* __restrict__ xh) {
    int idx = blockIdx.x * blockDim.x + threadIdx.x;
    if (idx >= TOK * C) return;
    int ww = idx % W;
    int hh = (idx / W) % H;
    int c  = (idx / (H * W)) % C;
    int b  = idx / (C * H * W);
    const float* xb = x + ((size_t)b * C + c) * H * W;
    float acc = 0.f;
#pragma unroll
    for (int dy = 0; dy < 3; dy++) {
        int h2 = hh + dy - 1;
        if (h2 < 0 || h2 >= H) continue;
#pragma unroll
        for (int dx = 0; dx < 3; dx++) {
            int w2 = ww + dx - 1;
            if (w2 < 0 || w2 >= W) continue;
            acc += xb[h2 * W + w2] * wt[c * 9 + dy * 3 + dx];
        }
    }
    size_t tokn = ((size_t)b * H + hh) * W + ww;
    xh[tokn * C + c] = xb[hh * W + ww] + acc + bias[c];
}

// LayerNorm over C (one warp per token) -> bf16
__global__ void ln_bf16_kernel(const float* __restrict__ xin, const float* __restrict__ g,
                               const float* __restrict__ b, __bf16* __restrict__ yout) {
    int warp = blockIdx.x * (blockDim.x >> 5) + (threadIdx.x >> 5);
    if (warp >= TOK) return;
    int lane = threadIdx.x & 31;
    const float* row = xin + (size_t)warp * C;
    float vals[4], s = 0.f, s2 = 0.f;
#pragma unroll
    for (int k = 0; k < 4; k++) { float v = row[lane + 32 * k]; vals[k] = v; s += v; s2 += v * v; }
#pragma unroll
    for (int off = 16; off > 0; off >>= 1) { s += __shfl_xor(s, off, 32); s2 += __shfl_xor(s2, off, 32); }
    float m = s * (1.0f / C);
    float var = s2 * (1.0f / C) - m * m;
    float inv = rsqrtf(var + 1e-6f);
#pragma unroll
    for (int k = 0; k < 4; k++) {
        int c = lane + 32 * k;
        yout[(size_t)warp * C + c] = (__bf16)((vals[k] - m) * inv * g[c] + b[c]);
    }
}

// ---------------- WMMA GEMM: out[M,N] = A[M,K]@B[K,N] + bias (+epilogue) ----------------
// Bt is transposed weights [N,K]. Each wave computes a 16x64 strip (4 WMMAs per K step
// sharing one A fragment).
// EPI 0: fp32 out + bias
// EPI 1: fp32 out + bias + residual
// EPI 2: bf16 out = gelu(acc + bias)
// EPI 3: fp32 NCHW out = acc + bias + residual (row->(b,h,w), col->channel)
template <int KDIM, int NDIM, int EPI>
__global__ void gemm_wmma_kernel(const __bf16* __restrict__ A, const __bf16* __restrict__ Bt,
                                 const float* __restrict__ bias, const float* __restrict__ res,
                                 void* __restrict__ outp) {
    constexpr int NG = NDIM / 64;
    constexpr int TILES = (TOK / 16) * NG;
    int wave = blockIdx.x * (blockDim.x >> 5) + (threadIdx.x >> 5);
    if (wave >= TILES) return;
    int mt = wave / NG, ng = wave % NG;
    int row0 = mt * 16, colg = ng * 64;
    int lane = threadIdx.x & 31;

    v8f acc[4] = {{}, {}, {}, {}};
#pragma unroll 2
    for (int k0 = 0; k0 < KDIM; k0 += 32) {
        v16bf a = frag_a_rm(A, KDIM, row0, k0);
#pragma unroll
        for (int j = 0; j < 4; j++) {
            v16bf bf = frag_b_t(Bt, KDIM, k0, colg + j * 16);
            acc[j] = __builtin_amdgcn_wmma_f32_16x16x32_bf16(false, a, false, bf, (short)0, acc[j],
                                                             false, false);
        }
    }
    int rbase = row0 + ((lane >> 4) << 3);
#pragma unroll
    for (int j = 0; j < 4; j++) {
        int col = colg + j * 16 + (lane & 15);
        float bv = bias[col];
#pragma unroll
        for (int i = 0; i < 8; i++) {
            int row = rbase + i;
            float v = acc[j][i] + bv;
            if constexpr (EPI == 0) {
                ((float*)outp)[(size_t)row * NDIM + col] = v;
            } else if constexpr (EPI == 1) {
                ((float*)outp)[(size_t)row * NDIM + col] = v + res[(size_t)row * NDIM + col];
            } else if constexpr (EPI == 2) {
                float gl = 0.5f * v * (1.0f + erff(v * 0.70710678118654752f));
                ((__bf16*)outp)[(size_t)row * NDIM + col] = (__bf16)gl;
            } else {
                float r = v + res[(size_t)row * C + col];
                int b = row / (H * W);
                int rem = row - b * (H * W);
                int hh = rem / W, ww = rem % W;
                ((float*)outp)[(((size_t)b * C + col) * H + hh) * W + ww] = r;
            }
        }
    }
}

// ---------------- window means for routing ----------------
__global__ void win_mean_kernel(const float* __restrict__ qkv, float* __restrict__ qwin,
                                float* __restrict__ kwin) {
    int blk = blockIdx.x;                 // b*P2 + p
    int p = blk % P2, b = blk / P2;
    int wy = p / NWIN, wx = p % NWIN;
    int c = threadIdx.x;                  // 0..127
    float qs = 0.f, ks = 0.f;
    for (int pix = 0; pix < W2; pix++) {
        int i = pix >> 3, j = pix & 7;
        size_t tokn = ((size_t)b * H + wy * WH + i) * W + wx * WH + j;
        qs += qkv[tokn * NQKV + c];
        ks += qkv[tokn * NQKV + QK + c];
    }
    qwin[(size_t)blk * C + c] = qs * (1.f / W2);
    kwin[(size_t)blk * C + c] = ks * (1.f / W2);
}

// ---------------- routing logits + top-4 ----------------
__global__ void routing_topk_kernel(const float* __restrict__ qwin, const float* __restrict__ kwin,
                                    int* __restrict__ idxb) {
    __shared__ float lg[P2];
    int blk = blockIdx.x;
    int b = blk / P2;
    int tid = threadIdx.x;                // 64 threads
    if (tid < P2) {
        const float* qr = qwin + (size_t)blk * C;
        const float* kr = kwin + ((size_t)b * P2 + tid) * C;
        float s = 0.f;
        for (int c = 0; c < C; c++) s += qr[c] * kr[c];
        lg[tid] = s * SCALE;
    }
    __syncthreads();
    if (tid == 0) {
        unsigned long long mask = 0;
        for (int s = 0; s < TOPK; s++) {
            float best = -1e30f; int bi = 0;
            for (int j = 0; j < P2; j++)
                if (!((mask >> j) & 1ull) && lg[j] > best) { best = lg[j]; bi = j; }
            mask |= 1ull << bi;
            idxb[(size_t)blk * TOPK + s] = bi;
        }
    }
}

// ---------------- per-(b, window, head) gathered attention ----------------
// q: 64x32, k/v: 256x32 (top-4 windows x 64 px). Stage q/k/vT in LDS as bf16 once,
// then all WMMA fragments are contiguous LDS loads. S/P softmax in LDS.
// Dynamic LDS layout:
//   S  : 64*256 f32   @ 0        (65536 B)
//   P  : 64*256 bf16  @ 65536    (32768 B)
//   Qs : 64*32  bf16  @ 98304    ( 4096 B)
//   Ks : 256*32 bf16  @ 102400   (16384 B)  (row = key pixel, col = d) == Bt for S
//   Vt : 32*256 bf16  @ 118784   (16384 B)  (row = d, col = key pixel) == Bt for O
constexpr size_t ATTN_LDS = 65536 + 32768 + 4096 + 16384 + 16384;
__global__ void attn_kernel(const float* __restrict__ qkv, const int* __restrict__ idxb,
                            float* __restrict__ oatt) {
    extern __shared__ char smem[];
    float*  S  = (float*)smem;
    __bf16* P  = (__bf16*)(smem + 65536);
    __bf16* Qs = (__bf16*)(smem + 98304);
    __bf16* Ks = (__bf16*)(smem + 102400);
    __bf16* Vt = (__bf16*)(smem + 118784);
    __shared__ int sel[TOPK];

    int blk = blockIdx.x;
    int head = blk % HEADS;
    int p = (blk / HEADS) % P2;
    int b = blk / (HEADS * P2);
    int wy = p / NWIN, wx = p % NWIN;
    int tid = threadIdx.x, wave = tid >> 5, lane = tid & 31;

    if (tid < TOPK) sel[tid] = idxb[(b * P2 + p) * TOPK + tid];
    __syncthreads();

    auto qtok = [&](int r) { int i = r >> 3, j = r & 7; return (b * H + wy * WH + i) * W + wx * WH + j; };
    auto ktok = [&](int n) {
        int s = n >> 6, pix = n & 63, kp = sel[s];
        int ky = kp / NWIN, kx = kp % NWIN, i = pix >> 3, j = pix & 7;
        return (b * H + ky * WH + i) * W + kx * WH + j;
    };

    // Phase 0: stage q, k, vT into LDS (bf16)
    {
        int r = tid >> 2;                 // q: 4 threads per row
        int d0 = (tid & 3) * 8;
        const float* src = qkv + (size_t)qtok(r) * NQKV + head * HD + d0;
        __bf16* dst = Qs + r * HD + d0;
#pragma unroll
        for (int i = 0; i < 8; i++) dst[i] = (__bf16)src[i];
    }
    {
        int kt = ktok(tid);               // k: one key pixel per thread
        const float* src = qkv + (size_t)kt * NQKV + QK + head * HD;
        __bf16* dst = Ks + tid * HD;
#pragma unroll
        for (int i = 0; i < HD; i++) dst[i] = (__bf16)src[i];
        const float* vsrc = qkv + (size_t)kt * NQKV + 2 * QK + head * HD;
#pragma unroll
        for (int d = 0; d < HD; d++) Vt[d * 256 + tid] = (__bf16)vsrc[d];
    }
    __syncthreads();

    // Phase 1: S = (q*SCALE) @ k^T. Wave w: mt = w>>1, 8 N-tiles, one A frag reused 8x.
    {
        int mt = wave >> 1;
        v16bf a = frag_a_rm(Qs, HD, mt * 16, 0);
        int ncol_lo = lane & 15;
        int rb = mt * 16 + ((lane >> 4) << 3);
#pragma unroll
        for (int j = 0; j < 8; j++) {
            int nt = (wave & 1) * 8 + j;
            v16bf bf = frag_b_t(Ks, HD, 0, nt * 16);
            v8f acc = {};
            acc = __builtin_amdgcn_wmma_f32_16x16x32_bf16(false, a, false, bf, (short)0, acc,
                                                          false, false);
            int ncol = nt * 16 + ncol_lo;
#pragma unroll
            for (int i = 0; i < 8; i++) S[(rb + i) * 256 + ncol] = acc[i] * SCALE;
        }
    }
    __syncthreads();

    // Phase 2: row softmax, 4 threads per row (each a contiguous 64-elem chunk),
    // reductions over the 4 sub-lanes via shfl_xor (sub-lanes are consecutive lanes).
    {
        int r = tid >> 2;                 // 0..63
        int sub = tid & 3;
        float* chunk = S + r * 256 + sub * 64;
        float mx = -1e30f;
#pragma unroll 4
        for (int k = 0; k < 64; k++) mx = fmaxf(mx, chunk[k]);
        mx = fmaxf(mx, __shfl_xor(mx, 1, 32));
        mx = fmaxf(mx, __shfl_xor(mx, 2, 32));
        float sum = 0.f;
#pragma unroll 4
        for (int k = 0; k < 64; k++) { float e = __expf(chunk[k] - mx); chunk[k] = e; sum += e; }
        sum += __shfl_xor(sum, 1, 32);
        sum += __shfl_xor(sum, 2, 32);
        float inv = 1.0f / sum;
        __bf16* pchunk = P + r * 256 + sub * 64;
#pragma unroll 4
        for (int k = 0; k < 64; k++) pchunk[k] = (__bf16)(chunk[k] * inv);
    }
    __syncthreads();

    // Phase 3: O = P @ V (64x32 via 4x2 tiles, one per wave), V fragments from Vt.
    {
        int mt = wave >> 1, nt = wave & 1;
        v8f acc = {};
#pragma unroll
        for (int k0 = 0; k0 < 256; k0 += 32) {
            v16bf a = frag_a_rm(P, 256, mt * 16, k0);
            v16bf bf = frag_b_t(Vt, 256, k0, nt * 16);
            acc = __builtin_amdgcn_wmma_f32_16x16x32_bf16(false, a, false, bf, (short)0, acc,
                                                          false, false);
        }
        int ch = head * HD + nt * 16 + (lane & 15);
        int rb = mt * 16 + ((lane >> 4) << 3);
#pragma unroll
        for (int i = 0; i < 8; i++) oatt[(size_t)qtok(rb + i) * C + ch] = acc[i];
    }
}

// ---------------- LEPE (5x5 dwconv of v) + attention out -> bf16 for Wo GEMM ----------------
__global__ void lepe_add_kernel(const float* __restrict__ qkv, const float* __restrict__ oatt,
                                const float* __restrict__ wt, const float* __restrict__ bias,
                                __bf16* __restrict__ t) {
    int idx = blockIdx.x * blockDim.x + threadIdx.x;
    if (idx >= TOK * C) return;
    int c = idx % C;
    int tokn = idx / C;
    int ww = tokn % W, hh = (tokn / W) % H, b = tokn / (H * W);
    float acc = bias[c];
#pragma unroll
    for (int dy = 0; dy < 5; dy++) {
        int h2 = hh + dy - 2;
        if (h2 < 0 || h2 >= H) continue;
#pragma unroll
        for (int dx = 0; dx < 5; dx++) {
            int w2 = ww + dx - 2;
            if (w2 < 0 || w2 >= W) continue;
            acc += qkv[(((size_t)b * H + h2) * W + w2) * NQKV + 2 * QK + c] * wt[c * 25 + dy * 5 + dx];
        }
    }
    t[idx] = (__bf16)(oatt[idx] + acc);
}

// ---------------- host side ----------------
extern "C" void kernel_launch(void* const* d_in, const int* in_sizes, int n_in,
                              void* d_out, int out_size, void* d_ws, size_t ws_size,
                              hipStream_t stream) {
    (void)in_sizes; (void)n_in; (void)out_size; (void)ws_size;
    const float* x      = (const float*)d_in[0];
    const float* pos_w  = (const float*)d_in[1];
    const float* pos_b  = (const float*)d_in[2];
    const float* ln1_g  = (const float*)d_in[3];
    const float* ln1_b  = (const float*)d_in[4];
    const float* qkv_w  = (const float*)d_in[5];
    const float* qkv_b  = (const float*)d_in[6];
    const float* lepe_w = (const float*)d_in[7];
    const float* lepe_b = (const float*)d_in[8];
    const float* wo_w   = (const float*)d_in[9];
    const float* wo_b   = (const float*)d_in[10];
    const float* ln2_g  = (const float*)d_in[11];
    const float* ln2_b  = (const float*)d_in[12];
    const float* mlp_w1 = (const float*)d_in[13];
    const float* mlp_b1 = (const float*)d_in[14];
    const float* mlp_w2 = (const float*)d_in[15];
    const float* mlp_b2 = (const float*)d_in[16];
    float* out = (float*)d_out;

    // workspace layout
    char* ws = (char*)d_ws;
    size_t off = 0;
    auto take = [&](size_t bytes) { char* p = ws + off; off += (bytes + 255) & ~(size_t)255; return p; };
    float*  xh    = (float*)take((size_t)TOK * C * 4);        // NHWC after pos conv
    float*  qkvb  = (float*)take((size_t)TOK * NQKV * 4);     // qkv fp32; later reused for MLP gelu buf
    float*  oatt  = (float*)take((size_t)TOK * C * 4);        // attention out; later reused as xh2
    __bf16* ybf   = (__bf16*)take((size_t)TOK * C * 2);       // LN1 out; later LN2 out
    __bf16* tbf   = (__bf16*)take((size_t)TOK * C * 2);       // (oatt + lepe) bf16
    float*  qwin  = (float*)take((size_t)BB * P2 * C * 4);
    float*  kwin  = (float*)take((size_t)BB * P2 * C * 4);
    int*    idxb  = (int*)take((size_t)BB * P2 * TOPK * 4);
    __bf16* wqkv  = (__bf16*)take((size_t)C * NQKV * 2);      // transposed [N,K]
    __bf16* wwo   = (__bf16*)take((size_t)C * C * 2);
    __bf16* wm1   = (__bf16*)take((size_t)C * 3 * C * 2);
    __bf16* wm2   = (__bf16*)take((size_t)3 * C * C * 2);
    __bf16* gbuf  = (__bf16*)qkvb;   // alias: qkv dead before MLP1
    float*  xh2   = oatt;            // alias: oatt dead before Wo GEMM writes

    const int TPB = 256;
    // 0) weights -> bf16 transposed [N,K]
    f2bf_t_kernel<<<(C * NQKV + TPB - 1) / TPB, TPB, 0, stream>>>(qkv_w, wqkv, C, NQKV);
    f2bf_t_kernel<<<(C * C + TPB - 1) / TPB, TPB, 0, stream>>>(wo_w, wwo, C, C);
    f2bf_t_kernel<<<(C * 3 * C + TPB - 1) / TPB, TPB, 0, stream>>>(mlp_w1, wm1, C, 3 * C);
    f2bf_t_kernel<<<(3 * C * C + TPB - 1) / TPB, TPB, 0, stream>>>(mlp_w2, wm2, 3 * C, C);

    // 1) pos dwconv + residual -> xh (NHWC fp32)
    posconv_kernel<<<(TOK * C + TPB - 1) / TPB, TPB, 0, stream>>>(x, pos_w, pos_b, xh);

    // 2) LN1 -> ybf
    ln_bf16_kernel<<<TOK / 8, TPB, 0, stream>>>(xh, ln1_g, ln1_b, ybf);

    // 3) QKV GEMM: [TOK,128]@[128,384] + bias -> qkvb fp32
    {
        constexpr int TILES = (TOK / 16) * (NQKV / 64);
        gemm_wmma_kernel<128, 384, 0><<<(TILES + 7) / 8, TPB, 0, stream>>>(ybf, wqkv, qkv_b, nullptr, qkvb);
    }

    // 4) routing: window means, logits, top-4
    win_mean_kernel<<<BB * P2, C, 0, stream>>>(qkvb, qwin, kwin);
    routing_topk_kernel<<<BB * P2, 64, 0, stream>>>(qwin, kwin, idxb);

    // 5) gathered attention per (b, window, head)
    attn_kernel<<<BB * P2 * HEADS, TPB, ATTN_LDS, stream>>>(qkvb, idxb, oatt);

    // 6) LEPE + add -> tbf (bf16)
    lepe_add_kernel<<<(TOK * C + TPB - 1) / TPB, TPB, 0, stream>>>(qkvb, oatt, lepe_w, lepe_b, tbf);

    // 7) Wo GEMM + residual: xh2 = xh + tbf@wo_w + wo_b
    {
        constexpr int TILES = (TOK / 16) * (C / 64);
        gemm_wmma_kernel<128, 128, 1><<<(TILES + 7) / 8, TPB, 0, stream>>>(tbf, wwo, wo_b, xh, xh2);
    }

    // 8) LN2 -> ybf (reused)
    ln_bf16_kernel<<<TOK / 8, TPB, 0, stream>>>(xh2, ln2_g, ln2_b, ybf);

    // 9) MLP1 + exact GELU -> gbuf (bf16)
    {
        constexpr int TILES = (TOK / 16) * (3 * C / 64);
        gemm_wmma_kernel<128, 384, 2><<<(TILES + 7) / 8, TPB, 0, stream>>>(ybf, wm1, mlp_b1, nullptr, gbuf);
    }

    // 10) MLP2 + residual, write NCHW output
    {
        constexpr int TILES = (TOK / 16) * (C / 64);
        gemm_wmma_kernel<384, 128, 3><<<(TILES + 7) / 8, TPB, 0, stream>>>(gbuf, wm2, mlp_b2, xh2, out);
    }
}
